// cpi_cpp_net_88244398063893
// MI455X (gfx1250) — compile-verified, matrix operand order checked by reading, not run on hardware
//
#include <hip/hip_runtime.h>
#include <hip/hip_bf16.h>

// ---------------------------------------------------------------------------
// CDNA5 / gfx1250 implementation of the CPI MPNN reference.
// Heavy GEMMs -> v_wmma_f32_16x16x32_bf16 (wave32, 16x16 tiles).
// ---------------------------------------------------------------------------

typedef __bf16 bf16_t;
typedef bf16_t v16bf __attribute__((ext_vector_type(16)));
typedef float  v8f   __attribute__((ext_vector_type(8)));

static constexpr int Nn = 32000;   // nodes
static constexpr int Ne = 64000;   // edges
static constexpr int Bg = 1280;    // graphs
static constexpr int Dd = 64;      // hidden dim
static constexpr int NPG = Nn / Bg; // 25 nodes per graph (sorted, contiguous)

__device__ __forceinline__ float sigf(float x) { return 1.f / (1.f + __expf(-x)); }

// ---------------------------------------------------------------------------
// bf16 16x32 fragment load, row-major source, per ISA 7.12.2:
// lane L<16  : row r0+L, K = {k0+0..7, k0+16..23}
// lane L>=16 : row r0+L-16, K = {k0+8..15, k0+24..31}
// B operand = W^T is loaded with the same layout from W (NOUT x K row-major),
// i.e. "row" is the output column.
// ---------------------------------------------------------------------------
__device__ __forceinline__ v16bf ld_frag16x32(const bf16_t* __restrict__ base,
                                              int ld, int r0, int k0) {
  int lane = threadIdx.x & 31;
  int row  = r0 + (lane & 15);
  const bf16_t* p = base + (size_t)row * ld + k0 + ((lane >> 4) << 3);
  v16bf f;
#pragma unroll
  for (int i = 0; i < 8; ++i) f[i] = p[i];
#pragma unroll
  for (int i = 0; i < 8; ++i) f[i + 8] = p[i + 16];
  return f;
}

// ---------------------------------------------------------------------------
// Generic WMMA GEMM: C[M x NOUT] = act( A[M x K] * W[NOUT x K]^T + bias (+C) )
// A, W bf16 row-major. M % 16 == 0, NOUT % 64 == 0, K % 32 == 0.
// One wave -> 16 rows x 64 cols (4 wmma accumulators). blockDim = 128.
// ---------------------------------------------------------------------------
template <int ACT, int OUTBF, int ACC>
__global__ void k_gemm(const bf16_t* __restrict__ A, int lda,
                       const bf16_t* __restrict__ W,
                       const float* __restrict__ bias,
                       void* __restrict__ Cv, int ldc,
                       int M, int NOUT, int K) {
  int gw = (int)((blockIdx.x * blockDim.x + threadIdx.x) >> 5);
  int mtiles = M >> 4;
  int nblocks = NOUT >> 6;
  if (gw >= mtiles * nblocks) return;
  int mt = gw % mtiles, nb = gw / mtiles;
  int m0 = mt << 4, n0 = nb << 6;
  int lane = threadIdx.x & 31;

  v8f acc[4] = {};
  for (int k0 = 0; k0 < K; k0 += 32) {
    v16bf a = ld_frag16x32(A, lda, m0, k0);
#pragma unroll
    for (int t = 0; t < 4; ++t) {
      v16bf b = ld_frag16x32(W, K, n0 + t * 16, k0);
      acc[t] = __builtin_amdgcn_wmma_f32_16x16x32_bf16(
          false, a, false, b, (short)0, acc[t], false, false);
    }
  }
  // D layout: VGPR r -> row m0 + r + 8*(lane>=16); col n0 + t*16 + (lane&15)
  int rowoff = (lane >> 4) << 3;
#pragma unroll
  for (int t = 0; t < 4; ++t) {
    int col = n0 + t * 16 + (lane & 15);
    float bv = bias ? bias[col] : 0.f;
#pragma unroll
    for (int r = 0; r < 8; ++r) {
      size_t idx = (size_t)(m0 + rowoff + r) * ldc + col;
      float v = acc[t][r] + bv;
      if (ACC) v += ((const float*)Cv)[idx];
      if (ACT) v = v > 0.f ? v : 0.f;
      if (OUTBF) ((bf16_t*)Cv)[idx] = (bf16_t)v;
      else       ((float*)Cv)[idx]  = v;
    }
  }
}

// ---------------------------------------------------------------------------
// fp32 -> bf16 convert with optional column padding (zeros)
// ---------------------------------------------------------------------------
__global__ void k_cvt_pad(const float* __restrict__ s, bf16_t* __restrict__ d,
                          int rows, int cin, int cout) {
  long i = (long)blockIdx.x * blockDim.x + threadIdx.x;
  if (i >= (long)rows * cout) return;
  int r = (int)(i / cout), c = (int)(i % cout);
  d[i] = (c < cin) ? (bf16_t)s[(size_t)r * cin + c] : (bf16_t)0.f;
}

__global__ void k_f32_to_bf16(const float* __restrict__ s,
                              bf16_t* __restrict__ d, long n) {
  long i = (long)blockIdx.x * blockDim.x + threadIdx.x;
  if (i < n) d[i] = (bf16_t)s[i];
}

// ---------------------------------------------------------------------------
// Per-edge message: msg = h[src] (1x64) @ ew[e] (64x64); atomic add into agg[dst]
// One wave per edge; lane l owns output cols l and l+32 (coalesced 64-wide rows)
// ---------------------------------------------------------------------------
__global__ void k_edge_msg(const bf16_t* __restrict__ hb,
                           const bf16_t* __restrict__ ew,
                           const int* __restrict__ src,
                           const int* __restrict__ dst,
                           float* __restrict__ agg, int E) {
  int wave = (int)((blockIdx.x * blockDim.x + threadIdx.x) >> 5);
  if (wave >= E) return;
  int lane = threadIdx.x & 31;
  const bf16_t* hrow = hb + (size_t)src[wave] * Dd;
  const bf16_t* Mrow = ew + (size_t)wave * (Dd * Dd);
  __builtin_prefetch(Mrow + 2048, 0, 0);   // global_prefetch_b8: second half of tile
  float a0 = 0.f, a1 = 0.f;
#pragma unroll 4
  for (int i = 0; i < Dd; ++i) {
    float hv = (float)hrow[i];
    a0 += hv * (float)Mrow[i * Dd + lane];
    a1 += hv * (float)Mrow[i * Dd + 32 + lane];
  }
  float* arow = agg + (size_t)dst[wave] * Dd;
  atomicAdd(arow + lane, a0);
  atomicAdd(arow + 32 + lane, a1);
}

// x = relu(agg + b_conv) -> bf16
__global__ void k_bias_relu_bf16(const float* __restrict__ agg,
                                 const float* __restrict__ b,
                                 bf16_t* __restrict__ xb, long n) {
  long i = (long)blockIdx.x * blockDim.x + threadIdx.x;
  if (i >= n) return;
  float v = agg[i] + b[i & (Dd - 1)];
  xb[i] = (bf16_t)(v > 0.f ? v : 0.f);
}

// GRU cell update (seq len 1)
__global__ void k_gru(const float* __restrict__ gi, const float* __restrict__ gh,
                      float* __restrict__ h, bf16_t* __restrict__ hb, long n) {
  long i = (long)blockIdx.x * blockDim.x + threadIdx.x;
  if (i >= n) return;
  int node = (int)(i >> 6), d = (int)(i & 63);
  const float* gir = gi + (size_t)node * 192;
  const float* ghr = gh + (size_t)node * 192;
  float r  = sigf(gir[d] + ghr[d]);
  float z  = sigf(gir[64 + d] + ghr[64 + d]);
  float nn = tanhf(gir[128 + d] + r * ghr[128 + d]);
  float hv = (1.f - z) * nn + z * h[i];
  h[i] = hv;
  hb[i] = (bf16_t)hv;
}

// LSTM cell (per layer)
__global__ void k_lstm_cell(const float* __restrict__ gates, float* __restrict__ c,
                            float* __restrict__ hs, bf16_t* __restrict__ hsb, long n) {
  long idx = (long)blockIdx.x * blockDim.x + threadIdx.x;
  if (idx >= n) return;
  int b = (int)(idx >> 6), d = (int)(idx & 63);
  const float* g = gates + (size_t)b * 256;
  float ig = sigf(g[d]);
  float fg = sigf(g[64 + d]);
  float gg = tanhf(g[128 + d]);
  float og = sigf(g[192 + d]);
  float cv = fg * c[idx] + ig * gg;
  c[idx] = cv;
  float hv = og * tanhf(cv);
  hs[idx] = hv;
  hsb[idx] = (bf16_t)hv;
}

// Set2Set attention readout: one block (64 threads) per graph (25 contiguous nodes)
__global__ void k_attention(const float* __restrict__ h, const float* __restrict__ q,
                            float* __restrict__ qstar, bf16_t* __restrict__ qstarb) {
  __shared__ float sE[32];
  __shared__ float sQ[64];
  int g = blockIdx.x;
  int t = threadIdx.x;
  sQ[t] = q[(size_t)g * Dd + t];
  __syncthreads();
  int base = g * NPG;
  if (t < NPG) {
    const float* hr = h + (size_t)(base + t) * Dd;
    float e = 0.f;
    for (int i = 0; i < Dd; ++i) e += hr[i] * sQ[i];
    sE[t] = e;
  }
  __syncthreads();
  if (t == 0) {
    float m = -1e30f;
    for (int k = 0; k < NPG; ++k) m = fmaxf(m, sE[k]);
    float s = 0.f;
    for (int k = 0; k < NPG; ++k) { float ex = __expf(sE[k] - m); sE[k] = ex; s += ex; }
    float inv = 1.f / s;
    for (int k = 0; k < NPG; ++k) sE[k] *= inv;
  }
  __syncthreads();
  float ro = 0.f;
  for (int k = 0; k < NPG; ++k) ro += sE[k] * h[(size_t)(base + k) * Dd + t];
  qstar[(size_t)g * 128 + t]       = sQ[t];
  qstar[(size_t)g * 128 + 64 + t]  = ro;
  qstarb[(size_t)g * 128 + t]      = (bf16_t)sQ[t];
  qstarb[(size_t)g * 128 + 64 + t] = (bf16_t)ro;
}

// concat [q_star | protein | 0-pad] -> bf16 (Bg x 256)
__global__ void k_concat(const float* __restrict__ qstar, const float* __restrict__ prot,
                         bf16_t* __restrict__ d, long n) {
  long i = (long)blockIdx.x * blockDim.x + threadIdx.x;
  if (i >= n) return;
  int r = (int)(i >> 8), c = (int)(i & 255);
  float v = (c < 128) ? qstar[(size_t)r * 128 + c]
          : (c < 228) ? prot[(size_t)r * 100 + (c - 128)]
          : 0.f;
  d[i] = (bf16_t)v;
}

// training-mode BatchNorm over batch; one block per column; emits bf16
__global__ void k_bn(const float* __restrict__ y, const float* __restrict__ g,
                     const float* __restrict__ beta, bf16_t* __restrict__ xb, int B_) {
  __shared__ float ssum[256];
  __shared__ float ssq[256];
  int c = blockIdx.x, t = threadIdx.x;
  float s = 0.f, sq = 0.f;
  for (int r = t; r < B_; r += 256) {
    float v = y[(size_t)r * 64 + c];
    s += v; sq += v * v;
  }
  ssum[t] = s; ssq[t] = sq;
  __syncthreads();
  for (int off = 128; off > 0; off >>= 1) {
    if (t < off) { ssum[t] += ssum[t + off]; ssq[t] += ssq[t + off]; }
    __syncthreads();
  }
  float mu = ssum[0] / B_;
  float var = ssq[0] / B_ - mu * mu;
  float scale = g[c] * rsqrtf(var + 1e-5f);
  float sh = beta[c] - mu * scale;
  for (int r = t; r < B_; r += 256)
    xb[(size_t)r * 64 + c] = (bf16_t)(y[(size_t)r * 64 + c] * scale + sh);
}

// final 64 -> 2 projection
__global__ void k_pred(const bf16_t* __restrict__ x, const float* __restrict__ W,
                       const float* __restrict__ b, float* __restrict__ out, int B_) {
  int i = blockIdx.x * blockDim.x + threadIdx.x;
  if (i >= B_ * 2) return;
  int r = i >> 1, j = i & 1;
  float s = b[j];
  const bf16_t* xr = x + (size_t)r * 64;
  const float* wr = W + j * 64;
  for (int k = 0; k < 64; ++k) s += (float)xr[k] * wr[k];
  out[i] = s;
}

// ---------------------------------------------------------------------------
// Host orchestration
// ---------------------------------------------------------------------------
static inline int cdiv(long a, long b) { return (int)((a + b - 1) / b); }

#define GEMM(ACT, OUTBF, ACC, A, lda, W, bias, C, ldc, M, NOUT, K)                \
  do {                                                                            \
    int waves_ = ((M) / 16) * ((NOUT) / 64);                                      \
    k_gemm<ACT, OUTBF, ACC><<<cdiv(waves_, 4), 128, 0, stream>>>(                 \
        (A), (lda), (W), (bias), (void*)(C), (ldc), (M), (NOUT), (K));            \
  } while (0)

extern "C" void kernel_launch(void* const* d_in, const int* in_sizes, int n_in,
                              void* d_out, int out_size, void* d_ws, size_t ws_size,
                              hipStream_t stream) {
  // ---- inputs (setup_inputs dict order, params flattened in dict/list order)
  const float* node_feats = (const float*)d_in[0];   // N x 74
  const float* edge_feats = (const float*)d_in[1];   // E x 12
  const float* prot       = (const float*)d_in[2];   // B x 100
  const int*   src        = (const int*)d_in[3];
  const int*   dst        = (const int*)d_in[4];
  // d_in[5] = node_graph (implicit: 25 contiguous nodes per graph)
  const float* Wp    = (const float*)d_in[6];   const float* bp    = (const float*)d_in[7];
  const float* We1   = (const float*)d_in[8];   const float* be1   = (const float*)d_in[9];
  const float* We2   = (const float*)d_in[10];  const float* be2   = (const float*)d_in[11];
  const float* bconv = (const float*)d_in[12];
  const float* W_ih  = (const float*)d_in[13];  const float* W_hh  = (const float*)d_in[14];
  const float* b_ih  = (const float*)d_in[15];  const float* b_hh  = (const float*)d_in[16];
  const float* Lih[3] = {(const float*)d_in[17], (const float*)d_in[18], (const float*)d_in[19]};
  const float* Lhh[3] = {(const float*)d_in[20], (const float*)d_in[21], (const float*)d_in[22]};
  const float* bih[3] = {(const float*)d_in[23], (const float*)d_in[24], (const float*)d_in[25]};
  const float* bhh[3] = {(const float*)d_in[26], (const float*)d_in[27], (const float*)d_in[28]};
  const float* W_m0 = (const float*)d_in[29]; const float* b_m0 = (const float*)d_in[30];
  const float* g_m0 = (const float*)d_in[31]; const float* be_m0 = (const float*)d_in[32];
  const float* W_m1 = (const float*)d_in[33]; const float* b_m1 = (const float*)d_in[34];
  const float* g_m1 = (const float*)d_in[35]; const float* be_m1 = (const float*)d_in[36];
  const float* W_pr = (const float*)d_in[37]; const float* b_pr = (const float*)d_in[38];
  float* out = (float*)d_out;                  // B x 2

  // ---- workspace arena
  char* wsb = (char*)d_ws;
  size_t off = 0;
  auto alloc = [&](size_t bytes) -> void* {
    void* p = wsb + off;
    off = (off + bytes + 255) & ~(size_t)255;
    return p;
  };
  // bf16 weight copies (K padded to mult of 32)
  bf16_t* wWp   = (bf16_t*)alloc((size_t)64 * 96 * 2);
  bf16_t* wWe1  = (bf16_t*)alloc((size_t)128 * 32 * 2);
  bf16_t* wWe2  = (bf16_t*)alloc((size_t)4096 * 128 * 2);
  bf16_t* wWih  = (bf16_t*)alloc((size_t)192 * 64 * 2);
  bf16_t* wWhh  = (bf16_t*)alloc((size_t)192 * 64 * 2);
  bf16_t* wLih0 = (bf16_t*)alloc((size_t)256 * 128 * 2);
  bf16_t* wLih1 = (bf16_t*)alloc((size_t)256 * 64 * 2);
  bf16_t* wLih2 = (bf16_t*)alloc((size_t)256 * 64 * 2);
  bf16_t* wLhh[3];
  for (int l = 0; l < 3; ++l) wLhh[l] = (bf16_t*)alloc((size_t)256 * 64 * 2);
  bf16_t* wWm0  = (bf16_t*)alloc((size_t)64 * 256 * 2);
  bf16_t* wWm1  = (bf16_t*)alloc((size_t)64 * 64 * 2);
  // activations
  bf16_t* Anode = (bf16_t*)alloc((size_t)Nn * 96 * 2);
  bf16_t* Aedge = (bf16_t*)alloc((size_t)Ne * 32 * 2);
  bf16_t* tEdge = (bf16_t*)alloc((size_t)Ne * 128 * 2);
  bf16_t* ew    = (bf16_t*)alloc((size_t)Ne * 4096 * 2);   // 512 MB
  float*  h     = (float*)alloc((size_t)Nn * Dd * 4);
  bf16_t* hbuf  = (bf16_t*)alloc((size_t)Nn * Dd * 2);
  float*  agg   = (float*)alloc((size_t)Nn * Dd * 4);
  bf16_t* xb    = (bf16_t*)alloc((size_t)Nn * Dd * 2);
  float*  gi    = (float*)alloc((size_t)Nn * 192 * 4);
  float*  gh    = (float*)alloc((size_t)Nn * 192 * 4);
  float*  hsAll = (float*)alloc((size_t)3 * Bg * Dd * 4);
  bf16_t* hsbAll= (bf16_t*)alloc((size_t)3 * Bg * Dd * 2);
  float*  csAll = (float*)alloc((size_t)3 * Bg * Dd * 4);
  float*  gates = (float*)alloc((size_t)Bg * 256 * 4);
  float*  qstar = (float*)alloc((size_t)Bg * 128 * 4);
  bf16_t* qstarb= (bf16_t*)alloc((size_t)Bg * 128 * 2);
  bf16_t* mlpin = (bf16_t*)alloc((size_t)Bg * 256 * 2);
  float*  y0    = (float*)alloc((size_t)Bg * 64 * 4);
  bf16_t* xb1   = (bf16_t*)alloc((size_t)Bg * 64 * 2);
  float*  y1    = (float*)alloc((size_t)Bg * 64 * 4);
  bf16_t* xb2   = (bf16_t*)alloc((size_t)Bg * 64 * 2);
  (void)ws_size; (void)in_sizes; (void)n_in; (void)out_size;

  auto cvt = [&](const float* s, bf16_t* d, int rows, int cin, int cout) {
    long n = (long)rows * cout;
    k_cvt_pad<<<cdiv(n, 256), 256, 0, stream>>>(s, d, rows, cin, cout);
  };

  // ---- weight conversion (fp32 -> bf16, zero-pad K)
  cvt(Wp, wWp, 64, 74, 96);
  cvt(We1, wWe1, 128, 12, 32);
  cvt(We2, wWe2, 4096, 128, 128);
  cvt(W_ih, wWih, 192, 64, 64);
  cvt(W_hh, wWhh, 192, 64, 64);
  cvt(Lih[0], wLih0, 256, 128, 128);
  cvt(Lih[1], wLih1, 256, 64, 64);
  cvt(Lih[2], wLih2, 256, 64, 64);
  for (int l = 0; l < 3; ++l) cvt(Lhh[l], wLhh[l], 256, 64, 64);
  cvt(W_m0, wWm0, 64, 228, 256);
  cvt(W_m1, wWm1, 64, 64, 64);
  // ---- input conversion
  cvt(node_feats, Anode, Nn, 74, 96);
  cvt(edge_feats, Aedge, Ne, 12, 32);

  // ---- node projection: h = relu(node @ Wp^T + bp)
  GEMM(1, 0, 0, Anode, 96, wWp, bp, h, 64, Nn, 64, 96);
  k_f32_to_bf16<<<cdiv((long)Nn * Dd, 256), 256, 0, stream>>>(h, hbuf, (long)Nn * Dd);

  // ---- edge network: ew = (relu(ef @ We1^T + be1)) @ We2^T + be2  (bf16, once)
  GEMM(1, 1, 0, Aedge, 32, wWe1, be1, tEdge, 128, Ne, 128, 32);
  GEMM(0, 1, 0, tEdge, 128, wWe2, be2, ew, 4096, Ne, 4096, 128);

  // ---- 6 message-passing steps with GRU
  for (int step = 0; step < 6; ++step) {
    hipMemsetAsync(agg, 0, (size_t)Nn * Dd * 4, stream);
    k_edge_msg<<<cdiv((long)Ne * 32, 256), 256, 0, stream>>>(hbuf, ew, src, dst, agg, Ne);
    k_bias_relu_bf16<<<cdiv((long)Nn * Dd, 256), 256, 0, stream>>>(agg, bconv, xb, (long)Nn * Dd);
    GEMM(0, 0, 0, xb,   64, wWih, b_ih, gi, 192, Nn, 192, 64);
    GEMM(0, 0, 0, hbuf, 64, wWhh, b_hh, gh, 192, Nn, 192, 64);
    k_gru<<<cdiv((long)Nn * Dd, 256), 256, 0, stream>>>(gi, gh, h, hbuf, (long)Nn * Dd);
  }

  // ---- Set2Set readout
  hipMemsetAsync(hsAll, 0, (size_t)3 * Bg * Dd * 4, stream);
  hipMemsetAsync(hsbAll, 0, (size_t)3 * Bg * Dd * 2, stream);
  hipMemsetAsync(csAll, 0, (size_t)3 * Bg * Dd * 4, stream);
  hipMemsetAsync(qstar, 0, (size_t)Bg * 128 * 4, stream);
  hipMemsetAsync(qstarb, 0, (size_t)Bg * 128 * 2, stream);
  bf16_t* wLihArr[3] = {wLih0, wLih1, wLih2};
  for (int it = 0; it < 6; ++it) {
    for (int l = 0; l < 3; ++l) {
      const bf16_t* xin = (l == 0) ? qstarb : (hsbAll + (size_t)(l - 1) * Bg * Dd);
      int K = (l == 0) ? 128 : 64;
      GEMM(0, 0, 0, xin, K, wLihArr[l], bih[l], gates, 256, Bg, 256, K);
      GEMM(0, 0, 1, hsbAll + (size_t)l * Bg * Dd, 64, wLhh[l], bhh[l], gates, 256, Bg, 256, 64);
      k_lstm_cell<<<cdiv((long)Bg * Dd, 256), 256, 0, stream>>>(
          gates, csAll + (size_t)l * Bg * Dd, hsAll + (size_t)l * Bg * Dd,
          hsbAll + (size_t)l * Bg * Dd, (long)Bg * Dd);
    }
    k_attention<<<Bg, 64, 0, stream>>>(h, hsAll + (size_t)2 * Bg * Dd, qstar, qstarb);
  }

  // ---- MLP predictor (BN training mode)
  k_concat<<<cdiv((long)Bg * 256, 256), 256, 0, stream>>>(qstar, prot, mlpin, (long)Bg * 256);
  GEMM(1, 0, 0, mlpin, 256, wWm0, b_m0, y0, 64, Bg, 64, 256);
  k_bn<<<64, 256, 0, stream>>>(y0, g_m0, be_m0, xb1, Bg);
  GEMM(1, 0, 0, xb1, 64, wWm1, b_m1, y1, 64, Bg, 64, 64);
  k_bn<<<64, 256, 0, stream>>>(y1, g_m1, be_m1, xb2, Bg);
  k_pred<<<cdiv((long)Bg * 2, 256), 256, 0, stream>>>(xb2, W_pr, b_pr, out, Bg);
}
